// GraphTransformerLayer_86474871537938
// MI455X (gfx1250) — matrix operand order repaired and checked by reading, not compiled, not run on hardware
//
#include <hip/hip_runtime.h>

// ---------------------------------------------------------------------------
// Graph transformer layer for MI455X (gfx1250, wave32, WMMA).
// All matmuls run through v_wmma_f32_16x16x32_bf16.
// GEMM: 64x128 block tile, 16x64 wave tile (4 WMMAs per A fragment),
// with global_prefetch_b8 ahead of the K loop.
// Attention uses a pre-fused (adjacency + mask) table so the score epilogue
// is a single fmac per element.
// ---------------------------------------------------------------------------

typedef __bf16 bf16_t;
typedef __attribute__((ext_vector_type(16))) __bf16 v16bf;
typedef __attribute__((ext_vector_type(8)))  __bf16 v8bf;
typedef __attribute__((ext_vector_type(4)))  __bf16 v4bf;
typedef __attribute__((ext_vector_type(8)))  float   v8f;
typedef __attribute__((ext_vector_type(4)))  float   v4f;

static constexpr int BB  = 8;
static constexpr int NN  = 1024;
static constexpr int DD  = 512;
static constexpr int HH  = 8;
static constexpr int FFN = 2048;
static constexpr int DH  = 64;
static constexpr int BN  = BB * NN;          // 8192 rows
static constexpr float EPSF  = 1e-5f;
static constexpr float NEGF  = -1.0e9f;
static constexpr float SCALE = 0.125f;       // 1/sqrt(64)

// ---------------------------------------------------------------------------
// WMMA helpers
// ---------------------------------------------------------------------------
__device__ __forceinline__ v8f wmma_bf16(v16bf a, v16bf b, v8f c) {
    return __builtin_amdgcn_wmma_f32_16x16x32_bf16(
        /*neg_a=*/false, a, /*neg_b=*/false, b,
        /*c_mod=*/(short)0, c, /*reuse_a=*/false, /*reuse_b=*/false);
}

// A fragment (16x32 bf16) from row-major A[.,lda].
// ISA layout: lanes 0-15 hold row (lane&15), K = 0..7 & 16..23;
//             lanes 16-31 same rows, K = 8..15 & 24..31.
__device__ __forceinline__ v16bf load_a_frag(const bf16_t* __restrict__ A,
                                             int lda, int rbase, int kbase, int lane) {
    const int row = rbase + (lane & 15);
    const int klo = (lane & 16) ? 8 : 0;
    const bf16_t* p = A + (size_t)row * lda + kbase + klo;
    v8bf lo = *(const v8bf*)p;
    v8bf hi = *(const v8bf*)(p + 16);
    v16bf a;
#pragma unroll
    for (int j = 0; j < 8; ++j) { a[j] = lo[j]; a[8 + j] = hi[j]; }
    return a;
}

// B fragment (32x16 bf16) where B[k][n] = W[nbase+n][kbase+k] (W row-major, ldw).
// ISA layout: lanes 0-15 hold column (lane&15), K = 0..7 & 16..23; lanes 16-31 K = 8..15 & 24..31.
__device__ __forceinline__ v16bf load_b_frag(const bf16_t* __restrict__ W,
                                             int ldw, int nbase, int kbase, int lane) {
    const int col = nbase + (lane & 15);
    const int kb  = (lane & 16) ? 8 : 0;
    const bf16_t* p = W + (size_t)col * ldw + kbase + kb;
    v8bf lo = *(const v8bf*)p;
    v8bf hi = *(const v8bf*)(p + 16);
    v16bf b;
#pragma unroll
    for (int j = 0; j < 8; ++j) { b[j] = lo[j]; b[8 + j] = hi[j]; }
    return b;
}

// ---------------------------------------------------------------------------
// fp32 -> bf16 conversion, vectorized (4 elements / thread)
// ---------------------------------------------------------------------------
__global__ void f2bf_kernel(const float* __restrict__ in, bf16_t* __restrict__ out, int n4) {
    int i = blockIdx.x * blockDim.x + threadIdx.x;
    if (i < n4) {
        v4f f = ((const v4f*)in)[i];
        v4bf o;
#pragma unroll
        for (int j = 0; j < 4; ++j) o[j] = (bf16_t)f[j];
        ((v4bf*)out)[i] = o;
    }
}

// ---------------------------------------------------------------------------
// Fuse adjacency + candidate mask once:  adjm = mask ? adj : -1e9
// (shared by all B*H = 64 attention passes)
// ---------------------------------------------------------------------------
__global__ void fuse_adj_mask_kernel(const float* __restrict__ adj,
                                     const unsigned char* __restrict__ mask,
                                     float* __restrict__ adjm, int n4) {
    int i = blockIdx.x * blockDim.x + threadIdx.x;
    if (i < n4) {
        v4f a = ((const v4f*)adj)[i];
        uchar4 m = ((const uchar4*)mask)[i];
        v4f o;
        o[0] = m.x ? a[0] : NEGF;
        o[1] = m.y ? a[1] : NEGF;
        o[2] = m.z ? a[2] : NEGF;
        o[3] = m.w ? a[3] : NEGF;
        ((v4f*)adjm)[i] = o;
    }
}

// ---------------------------------------------------------------------------
// Generic GEMM: C[M,Nc] = A[M,K] * W[Nc,K]^T  (torch-Linear convention).
// 256 threads = 8 waves; block tile 64(M) x 128(N); wave tile 16 x 64.
// Each A fragment feeds 4 WMMAs. Next K tiles prefetched via
// global_prefetch_b8 (CDNA5).
// EPI: 0 = f32 row-major out
//      1 = bf16, permute [BN,D] -> q/k layout [b,h,n,dh]
//      2 = bf16, permute [BN,D] -> vT layout  [b,h,dh,n]
//      3 = +bias, ReLU, bf16 row-major out
//      4 = +bias, f32 row-major out
// ---------------------------------------------------------------------------
template <int EPI>
__global__ void gemm_bf16_kernel(const bf16_t* __restrict__ A,
                                 const bf16_t* __restrict__ W,
                                 int M, int Nc, int K,
                                 float* __restrict__ outF,
                                 bf16_t* __restrict__ outB,
                                 const float* __restrict__ bias) {
    const int lane = threadIdx.x & 31;
    const int w    = threadIdx.x >> 5;          // 0..7
    const int mbase = blockIdx.x * 64  + (w & 3) * 16;
    const int nbase = blockIdx.y * 128 + (w >> 2) * 64;

    v8f acc0 = {}, acc1 = {}, acc2 = {}, acc3 = {};
    for (int k0 = 0; k0 < K; k0 += 32) {
        // prefetch two K-steps ahead (global_prefetch_b8)
        if (k0 + 64 < K) {
            __builtin_prefetch(A + (size_t)(mbase + (lane & 15)) * K + k0 + 64, 0, 3);
            __builtin_prefetch(W + (size_t)(nbase + (lane & 15)) * K + k0 + 64, 0, 3);
            __builtin_prefetch(W + (size_t)(nbase + 32 + (lane & 15)) * K + k0 + 64, 0, 3);
        }
        v16bf a  = load_a_frag(A, K, mbase, k0, lane);
        v16bf b0 = load_b_frag(W, K, nbase,      k0, lane);
        v16bf b1 = load_b_frag(W, K, nbase + 16, k0, lane);
        v16bf b2 = load_b_frag(W, K, nbase + 32, k0, lane);
        v16bf b3 = load_b_frag(W, K, nbase + 48, k0, lane);
        acc0 = wmma_bf16(a, b0, acc0);
        acc1 = wmma_bf16(a, b1, acc1);
        acc2 = wmma_bf16(a, b2, acc2);
        acc3 = wmma_bf16(a, b3, acc3);
    }

    const int rowoff = (lane & 16) ? 8 : 0;
    const int coll   = lane & 15;
#pragma unroll
    for (int t = 0; t < 4; ++t) {
        const v8f& acc = (t == 0) ? acc0 : (t == 1) ? acc1 : (t == 2) ? acc2 : acc3;
#pragma unroll
        for (int r = 0; r < 8; ++r) {
            const int m   = mbase + r + rowoff;
            const int col = nbase + t * 16 + coll;
            float v = acc[r];
            if constexpr (EPI == 0) {
                outF[(size_t)m * Nc + col] = v;
            } else if constexpr (EPI == 1) {
                const int b = m >> 10, n = m & 1023;
                const int h = col >> 6, dh = col & 63;
                outB[(((size_t)(b * HH + h)) * NN + n) * DH + dh] = (bf16_t)v;
            } else if constexpr (EPI == 2) {
                const int b = m >> 10, n = m & 1023;
                const int h = col >> 6, dh = col & 63;
                outB[(((size_t)(b * HH + h)) * DH + dh) * NN + n] = (bf16_t)v;
            } else if constexpr (EPI == 3) {
                v += bias[col];
                v = v > 0.f ? v : 0.f;
                outB[(size_t)m * Nc + col] = (bf16_t)v;
            } else { // EPI == 4
                v += bias[col];
                outF[(size_t)m * Nc + col] = v;
            }
        }
    }
}

// ---------------------------------------------------------------------------
// Fused attention per (b, 16-row q tile). 128 threads = 4 waves.
// Scores -> LDS (16x1024 f32, 64KB), softmax, mean accumulation into d_out,
// then attn @ V via WMMA into bf16 ctx [b, n, h*64+dh].
// adjm already holds (mask ? adj : -1e9), so score epilogue = fmac + store.
// ---------------------------------------------------------------------------
__global__ void attn_kernel(const bf16_t* __restrict__ q,
                            const bf16_t* __restrict__ k,
                            const bf16_t* __restrict__ vt,
                            const float*  __restrict__ adjm,
                            bf16_t* __restrict__ ctx,
                            float* __restrict__ gmean) {
    __shared__ float sc[16 * 1024];             // exactly 64 KB

    const int lane = threadIdx.x & 31;
    const int w    = threadIdx.x >> 5;          // 0..3
    const int b    = blockIdx.x >> 6;           // NN/16 = 64 q-tiles
    const int qt   = blockIdx.x & 63;
    const int q0   = qt * 16;

    for (int h = 0; h < HH; ++h) {
        const bf16_t* qh = q  + ((size_t)(b * HH + h)) * NN * DH;
        const bf16_t* kh = k  + ((size_t)(b * HH + h)) * NN * DH;
        const bf16_t* vh = vt + ((size_t)(b * HH + h)) * DH * NN;

        // Q fragments for this wave (full Dh=64: two K-halves)
        v16bf a0 = load_a_frag(qh, DH, q0, 0,  lane);
        v16bf a1 = load_a_frag(qh, DH, q0, 32, lane);

        // ---- scores: each wave does 16 of the 64 key tiles ----
        for (int kt = w; kt < 64; kt += 4) {
            v8f c = {};
            v16bf b0 = load_b_frag(kh, DH, kt * 16, 0,  lane);
            v16bf b1 = load_b_frag(kh, DH, kt * 16, 32, lane);
            c = wmma_bf16(a0, b0, c);
            c = wmma_bf16(a1, b1, c);

            const int rowoff = (lane & 16) ? 8 : 0;
            const int col    = kt * 16 + (lane & 15);
#pragma unroll
            for (int r = 0; r < 8; ++r) {
                const int lr = r + rowoff;
                const int qg = q0 + lr;
                sc[lr * 1024 + col] = fmaf(c[r], SCALE, adjm[(size_t)qg * NN + col]);
            }
        }
        __syncthreads();

        // ---- softmax + head-mean: wave w owns rows 4w..4w+3 ----
        for (int rr = 0; rr < 4; ++rr) {
            const int lr = w * 4 + rr;
            float* row = sc + lr * 1024;
            float mx = -3.0e38f;
            for (int j = lane; j < 1024; j += 32) mx = fmaxf(mx, row[j]);
#pragma unroll
            for (int o = 16; o > 0; o >>= 1) mx = fmaxf(mx, __shfl_xor(mx, o, 32));
            float sum = 0.f;
            for (int j = lane; j < 1024; j += 32) {
                float e = __expf(row[j] - mx);
                row[j] = e;
                sum += e;
            }
#pragma unroll
            for (int o = 16; o > 0; o >>= 1) sum += __shfl_xor(sum, o, 32);
            const float inv = 1.0f / sum;
            float* gm = gmean + ((size_t)b * NN + q0 + lr) * NN;
            for (int j = lane; j < 1024; j += 32) {
                float a = row[j] * inv;
                row[j] = a;
                float contrib = a * 0.125f;     // mean over 8 heads
                if (h == 0) gm[j] = contrib; else gm[j] += contrib;
            }
        }
        __syncthreads();

        // ---- attn @ V: wave w owns 16-wide d-column tile ----
        {
            const int dbase = w * 16;
            v8f c = {};
            for (int kt = 0; kt < 32; ++kt) {   // K = 1024 in steps of 32
                // A fragment from LDS attn (f32 -> bf16)
                const int arow = lane & 15;
                const int klo  = (lane & 16) ? 8 : 0;
                const float* p = sc + arow * 1024 + kt * 32 + klo;
                v16bf a;
#pragma unroll
                for (int j = 0; j < 8; ++j) {
                    a[j]     = (bf16_t)p[j];
                    a[8 + j] = (bf16_t)p[16 + j];
                }
                v16bf bb = load_b_frag(vh, NN, dbase, kt * 32, lane);
                c = wmma_bf16(a, bb, c);
            }
            const int rowoff = (lane & 16) ? 8 : 0;
            const int coll   = lane & 15;
#pragma unroll
            for (int r = 0; r < 8; ++r) {
                const int n    = q0 + r + rowoff;
                const int dcol = h * DH + dbase + coll;
                ctx[((size_t)b * NN + n) * DD + dcol] = (bf16_t)c[r];
            }
        }
        __syncthreads();
    }
}

// ---------------------------------------------------------------------------
// Residual + LayerNorm. One block (256 threads) per row, D = 512.
// Writes f32 z (for residual-2) and bf16 z (for next GEMM).
// ---------------------------------------------------------------------------
__global__ void ln_kernel(const float* __restrict__ xin,
                          const float* __restrict__ res,
                          const float* __restrict__ g,
                          const float* __restrict__ beta,
                          float* __restrict__ zf,          // may be null
                          bf16_t* __restrict__ zb) {       // may be null
    const int row = blockIdx.x;
    const int tid = threadIdx.x;
    const float* xr = xin + (size_t)row * DD;
    const float* ar = res + (size_t)row * DD;

    float v0 = xr[tid] + ar[tid];
    float v1 = xr[tid + 256] + ar[tid + 256];

    float s  = v0 + v1;
    float s2 = v0 * v0 + v1 * v1;
#pragma unroll
    for (int o = 16; o > 0; o >>= 1) {
        s  += __shfl_xor(s,  o, 32);
        s2 += __shfl_xor(s2, o, 32);
    }
    __shared__ float red[16];
    const int wv = tid >> 5, lane = tid & 31;
    if (lane == 0) { red[wv] = s; red[wv + 8] = s2; }
    __syncthreads();
    if (tid == 0) {
        float ts = 0.f, t2 = 0.f;
        for (int i = 0; i < 8; ++i) { ts += red[i]; t2 += red[i + 8]; }
        red[0] = ts; red[8] = t2;
    }
    __syncthreads();
    const float mu  = red[0] * (1.0f / DD);
    const float var = red[8] * (1.0f / DD) - mu * mu;
    const float rs  = rsqrtf(var + EPSF);

    float o0 = (v0 - mu) * rs * g[tid]       + beta[tid];
    float o1 = (v1 - mu) * rs * g[tid + 256] + beta[tid + 256];
    if (zf) { zf[(size_t)row * DD + tid] = o0; zf[(size_t)row * DD + tid + 256] = o1; }
    if (zb) { zb[(size_t)row * DD + tid] = (bf16_t)o0; zb[(size_t)row * DD + tid + 256] = (bf16_t)o1; }
}

// ---------------------------------------------------------------------------
// Host-side orchestration
// ---------------------------------------------------------------------------
extern "C" void kernel_launch(void* const* d_in, const int* in_sizes, int n_in,
                              void* d_out, int out_size, void* d_ws, size_t ws_size,
                              hipStream_t stream) {
    const float*         x    = (const float*)d_in[0];
    const float*         adj  = (const float*)d_in[1];
    const unsigned char* mask = (const unsigned char*)d_in[2];   // jnp.bool_ = 1 byte
    const float* Wq = (const float*)d_in[3];
    const float* Wk = (const float*)d_in[4];
    const float* Wv = (const float*)d_in[5];
    const float* Wo = (const float*)d_in[6];
    const float* W1 = (const float*)d_in[7];
    const float* b1 = (const float*)d_in[8];
    const float* W2 = (const float*)d_in[9];
    const float* b2 = (const float*)d_in[10];
    const float* g1 = (const float*)d_in[11];
    const float* be1= (const float*)d_in[12];
    const float* g2 = (const float*)d_in[13];
    const float* be2= (const float*)d_in[14];

    float* out   = (float*)d_out;                 // [B,N,D]
    float* gmean = out + (size_t)BB * NN * DD;    // [B,N,N]

    // workspace carve-up (~128 MB total), 256B aligned
    char* p = (char*)d_ws;
    auto alloc = [&](size_t bytes) -> void* {
        void* r = (void*)p;
        p += (bytes + 255) & ~(size_t)255;
        return r;
    };
    bf16_t* xb   = (bf16_t*)alloc((size_t)BN * DD * 2);
    bf16_t* wqb  = (bf16_t*)alloc((size_t)DD * DD * 2);
    bf16_t* wkb  = (bf16_t*)alloc((size_t)DD * DD * 2);
    bf16_t* wvb  = (bf16_t*)alloc((size_t)DD * DD * 2);
    bf16_t* wob  = (bf16_t*)alloc((size_t)DD * DD * 2);
    bf16_t* w1b  = (bf16_t*)alloc((size_t)FFN * DD * 2);
    bf16_t* w2b  = (bf16_t*)alloc((size_t)DD * FFN * 2);
    bf16_t* qb   = (bf16_t*)alloc((size_t)BB * HH * NN * DH * 2);
    bf16_t* kb   = (bf16_t*)alloc((size_t)BB * HH * NN * DH * 2);
    bf16_t* vtb  = (bf16_t*)alloc((size_t)BB * HH * DH * NN * 2);
    bf16_t* ctxb = (bf16_t*)alloc((size_t)BN * DD * 2);
    float*  adjm = (float*) alloc((size_t)NN * NN * 4);   // fused adj+mask
    float*  ao   = (float*) alloc((size_t)BN * DD * 4);   // attn_out
    float*  zf   = (float*) alloc((size_t)BN * DD * 4);   // z (f32)
    bf16_t* zb   = (bf16_t*)alloc((size_t)BN * DD * 2);   // z (bf16)
    bf16_t* hb   = (bf16_t*)alloc((size_t)BN * FFN * 2);  // relu hidden
    float*  zffn = (float*) alloc((size_t)BN * DD * 4);

    auto cvt = [&](const float* src, bf16_t* dst, int n) {
        const int n4 = n / 4;
        f2bf_kernel<<<(n4 + 255) / 256, 256, 0, stream>>>(src, dst, n4);
    };
    cvt(x,  xb,  BN * DD);
    cvt(Wq, wqb, DD * DD);
    cvt(Wk, wkb, DD * DD);
    cvt(Wv, wvb, DD * DD);
    cvt(Wo, wob, DD * DD);
    cvt(W1, w1b, FFN * DD);
    cvt(W2, w2b, DD * FFN);

    // fused adjacency+mask table (reused by all 64 (b,h) attention passes)
    {
        const int n4 = (NN * NN) / 4;
        fuse_adj_mask_kernel<<<(n4 + 255) / 256, 256, 0, stream>>>(adj, mask, adjm, n4);
    }

    // Q/K/V projections (x @ W^T), writing attention-friendly layouts
    dim3 gqkv(BN / 64, DD / 128);
    gemm_bf16_kernel<1><<<gqkv, 256, 0, stream>>>(xb, wqb, BN, DD, DD, nullptr, qb,  nullptr);
    gemm_bf16_kernel<1><<<gqkv, 256, 0, stream>>>(xb, wkb, BN, DD, DD, nullptr, kb,  nullptr);
    gemm_bf16_kernel<2><<<gqkv, 256, 0, stream>>>(xb, wvb, BN, DD, DD, nullptr, vtb, nullptr);

    // fused masked attention + softmax + head-mean + attn@V
    attn_kernel<<<BB * (NN / 16), 128, 0, stream>>>(qb, kb, vtb, adjm, ctxb, gmean);

    // output projection
    gemm_bf16_kernel<0><<<dim3(BN / 64, DD / 128), 256, 0, stream>>>(
        ctxb, wob, BN, DD, DD, ao, nullptr, nullptr);

    // z = LN(x + attn_out)
    ln_kernel<<<BN, 256, 0, stream>>>(x, ao, g1, be1, zf, zb);

    // h = relu(z @ W1^T + b1)
    gemm_bf16_kernel<3><<<dim3(BN / 64, FFN / 128), 256, 0, stream>>>(
        zb, w1b, BN, FFN, DD, nullptr, hb, b1);

    // z_ffn = h @ W2^T + b2
    gemm_bf16_kernel<4><<<dim3(BN / 64, DD / 128), 256, 0, stream>>>(
        hb, w2b, BN, DD, FFN, zffn, nullptr, b2);

    // out = LN(z + z_ffn)
    ln_kernel<<<BN, 256, 0, stream>>>(zf, zffn, g2, be2, out, nullptr);

    (void)in_sizes; (void)n_in; (void)out_size; (void)ws_size;
}
